// InverseFoldingTransformer_83038897701231
// MI455X (gfx1250) — matrix operand order repaired and checked by reading, not compiled, not run on hardware
//
#include <hip/hip_runtime.h>
#include <hip/hip_bf16.h>

// MI455X / gfx1250 implementation.
// All dense GEMMs go through v_wmma_f32_16x16x32_f16 (wave32, 16x16 tiles,
// f16 A/B, f32 accumulate). Activations keep an f16 shadow copy that feeds
// the next GEMM's A operand; f32 weights are converted to f16 while staging
// into LDS. GEMM staging uses GLOBAL_LOAD_ASYNC_TO_LDS_B128 (ASYNCcnt) with
// a double-buffered K loop when the toolchain exposes the builtins.
// Irregular sparse cross-attention (32 neighbors/token) runs on VALU with
// wave shuffles (it is ~0.1% of total FLOPs).

typedef __attribute__((ext_vector_type(16))) _Float16 v16h;
typedef __attribute__((ext_vector_type(8)))  float    v8f;
typedef __attribute__((ext_vector_type(4)))  int      i4x;

static constexpr int Dm  = 512;
static constexpr int Hh  = 8;
static constexpr int DH  = 64;
static constexpr int Ff  = 2048;
static constexpr int NLn = 6;
static constexpr int Vv  = 23;
static constexpr int Ll  = 1024;
static constexpr int NAGc = 8192;
static constexpr int KNBc = 32;

static constexpr int BM = 128, BN = 128, BK = 32;
static constexpr int AST = BK + 8;   // LDS row stride (halves) to spread banks
static constexpr int BST = BK + 8;

// ---- gfx1250 async global->LDS (ASYNCcnt) --------------------------------
#if defined(__has_builtin)
#  if __has_builtin(__builtin_amdgcn_global_load_async_to_lds_b128) && \
      __has_builtin(__builtin_amdgcn_s_wait_asynccnt)
#    define HAVE_ASYNC_LDS 1
#  endif
#endif
#ifndef HAVE_ASYNC_LDS
#  define HAVE_ASYNC_LDS 0
#endif

// Address-space-qualified int4 types (typedef form binds AS to the type).
typedef __attribute__((address_space(1))) i4x gi4;   // global int4
typedef __attribute__((address_space(3))) i4x li4;   // LDS int4

// Generic global pointer -> AS(1) int4* (same 64-bit address).
static __device__ __forceinline__ gi4* glob_i4(const void* p) {
  return (gi4*)(unsigned long long)p;
}
// Generic LDS pointer -> AS(3) int4*: ISA 10.2, generic LDS addr[31:0] is the
// LDS byte offset, so truncation yields the 32-bit LDS address.
static __device__ __forceinline__ li4* lds_i4(void* p) {
  return (li4*)(unsigned int)(unsigned long long)p;
}

static __device__ __forceinline__ void async_wait_all() {
#if HAVE_ASYNC_LDS
  __builtin_amdgcn_s_wait_asynccnt(0);
#endif
}

// Copy 32 bytes (16 halves) global -> LDS, fast path.
static __device__ __forceinline__ void stage16h(const _Float16* src, _Float16* dst) {
#if HAVE_ASYNC_LDS
  gi4* g = glob_i4(src);
  li4* l = lds_i4(dst);
  __builtin_amdgcn_global_load_async_to_lds_b128(g, l, 0, 0);
  __builtin_amdgcn_global_load_async_to_lds_b128(g, l, 16, 0);
#else
  ((uint4*)dst)[0] = ((const uint4*)src)[0];
  ((uint4*)dst)[1] = ((const uint4*)src)[1];
#endif
}

__device__ __forceinline__ float gelu_f(float x) {
  float x3 = x * x * x;
  return 0.5f * x * (1.0f + tanhf(0.7978845608028654f * (x + 0.044715f * x3)));
}

// ---------------------------------------------------------------------------
// Generic WMMA GEMM:  C[M,N] = act(scale * A[M,K] @ B + bias) + resid
//   A: f16 row-major (lda), per-batch offset batA (elements)
//   B: if BT==0: row-major [K,N] (f32 if !B_F16 else f16)
//      if BT==1: row-major [N,K] f16 (i.e. computes A @ B^T)
//   C: optional f32 (Cf) and/or f16 (Ch) outputs, ldc, per-batch batC
// Double-buffered K loop; async LDS staging when available.
// ---------------------------------------------------------------------------
template<bool B_F16, bool BT, bool GELU>
__global__ __launch_bounds__(256) void gemm_wmma_kernel(
    const _Float16* __restrict__ A, long lda, long batA,
    const void* __restrict__ Bv, long ldb, long batB,
    float* __restrict__ Cf, _Float16* __restrict__ Ch, long ldc, long batC,
    const float* __restrict__ bias, const float* __restrict__ resid,
    int M, int N, int K, float scale)
{
  __shared__ __align__(16) _Float16 Ash[2][BM * AST];
  __shared__ __align__(16) _Float16 Bsh[2][BN * BST];

  const int  tid  = threadIdx.x;
  const int  lane = tid & 31;
  const int  wid  = tid >> 5;       // 8 waves
  const int  wm   = wid & 3;        // 4 wave rows  -> 32 rows each
  const int  wn   = wid >> 2;       // 2 wave cols  -> 64 cols each
  const int  bz   = blockIdx.z;
  const long r0   = (long)blockIdx.y * BM;
  const long c0   = (long)blockIdx.x * BN;

  const _Float16* Ab = A + (long)bz * batA;
  const float*    Bf = (const float*)Bv;
  const _Float16* Bh = (const _Float16*)Bv;
  if (B_F16) Bh += (long)bz * batB; else Bf += (long)bz * batB;

  const v8f vzero = {0.f,0.f,0.f,0.f,0.f,0.f,0.f,0.f};
  v8f acc[2][4];
  for (int i = 0; i < 2; ++i)
    for (int j = 0; j < 4; ++j) acc[i][j] = vzero;

  const int srow = tid >> 1;            // 0..127 (tile row)
  const int scol = (tid & 1) * 16;      // 0 / 16 (half of K slice)

  // ---- tile staging (issues async loads on the fast path) ----
  auto stageA = [&](int k0, _Float16* abuf) {
    _Float16* dst = &abuf[srow * AST + scol];
    long gr = r0 + srow;
    int  kb = k0 + scol;
    int  krem = K - kb;
    if (gr < M && krem > 0) {
      const _Float16* src = Ab + gr * lda + kb;
      if (krem >= 16 && ((((unsigned long)src) & 15) == 0)) {
        stage16h(src, dst);
      } else {
        for (int i = 0; i < 16; ++i) dst[i] = (i < krem) ? src[i] : (_Float16)0.f;
      }
      if (k0 + BK < K) __builtin_prefetch(src + BK, 0, 1);
    } else {
      for (int i = 0; i < 16; ++i) dst[i] = (_Float16)0.f;
    }
  };
  auto stageB = [&](int k0, _Float16* bbuf) {
    _Float16* dst = &bbuf[srow * BST + scol];
    long gn = c0 + srow;
    int  kb = k0 + scol;
    int  krem = K - kb;
    if (BT) {
      if (gn < N && krem > 0) {
        const _Float16* src = Bh + gn * ldb + kb;
        if (krem >= 16 && ((((unsigned long)src) & 15) == 0)) {
          stage16h(src, dst);
        } else {
          for (int i = 0; i < 16; ++i) dst[i] = (i < krem) ? src[i] : (_Float16)0.f;
        }
      } else {
        for (int i = 0; i < 16; ++i) dst[i] = (_Float16)0.f;
      }
    } else {
      if (gn < N) {
        for (int i = 0; i < 16; ++i)
          dst[i] = (i < krem) ? (_Float16)(B_F16 ? (float)Bh[(long)(kb+i)*ldb + gn]
                                                 : Bf[(long)(kb+i)*ldb + gn])
                              : (_Float16)0.f;
      } else {
        for (int i = 0; i < 16; ++i) dst[i] = (_Float16)0.f;
      }
    }
  };

  // prologue: fill buffer 0
  stageA(0, Ash[0]);
  stageB(0, Bsh[0]);
  async_wait_all();
  __syncthreads();

  const int ml   = lane & 15;
  const int a_kb = (lane >> 4) << 3;    // lanes 16-31 start at K=8 (pairs)
  const int b_ks = (lane >> 4) << 3;    // lanes 16-31 start at K=8 (same as A)

  int buf = 0;
  for (int k0 = 0; k0 < K; k0 += BK) {
    // stage next K tile into the other buffer while we compute on this one
    int nk = k0 + BK;
    if (nk < K) {
      stageA(nk, Ash[buf ^ 1]);
      stageB(nk, Bsh[buf ^ 1]);
    }

    // ---- WMMA on current buffer: fragment layouts per CDNA5 ISA 7.12.2 ----
    const _Float16* Acur = Ash[buf];
    const _Float16* Bcur = Bsh[buf];
    for (int i = 0; i < 2; ++i) {
      union { v16h v; uint4 q[2]; } af;
      const _Float16* ap = &Acur[(wm*32 + i*16 + ml) * AST + a_kb];
      af.q[0] = *(const uint4*)ap;         // K = a_kb..a_kb+7
      af.q[1] = *(const uint4*)(ap + 16);  // K = 16+a_kb..16+a_kb+7
      for (int j = 0; j < 4; ++j) {
        union { v16h v; uint4 q[2]; } bfr;
        const _Float16* bp = &Bcur[(wn*64 + j*16 + ml) * BST + b_ks];
        bfr.q[0] = *(const uint4*)bp;        // K = b_ks..b_ks+7
        bfr.q[1] = *(const uint4*)(bp + 16); // K = 16+b_ks..16+b_ks+7
        acc[i][j] = __builtin_amdgcn_wmma_f32_16x16x32_f16(
            false, af.v, false, bfr.v, (short)0, acc[i][j], false, false);
      }
    }

    async_wait_all();   // next-buffer async stores to LDS complete (this wave)
    __syncthreads();    // all waves done reading cur / writing next
    buf ^= 1;
  }

  // ---- epilogue: C/D layout => lanes 0-15: M=r, lanes 16-31: M=8+r ----
  float*    Cfb = Cf ? (Cf + (long)bz * batC) : nullptr;
  _Float16* Chb = Ch ? (Ch + (long)bz * batC) : nullptr;
  const int rsub = (lane >> 4) * 8;
  const int csub = lane & 15;
  for (int i = 0; i < 2; ++i) {
    for (int j = 0; j < 4; ++j) {
      long c = c0 + wn*64 + j*16 + csub;
      if (c >= N) continue;
      long rb = r0 + wm*32 + i*16 + rsub;
      for (int r = 0; r < 8; ++r) {
        long rr = rb + r;
        if (rr >= M) break;
        float v = acc[i][j][r] * scale;
        if (bias)  v += bias[c];
        if (GELU)  v = gelu_f(v);
        if (resid) v += resid[rr * ldc + c];
        if (Cfb) Cfb[rr * ldc + c] = v;
        if (Chb) Chb[rr * ldc + c] = (_Float16)v;
      }
    }
  }
}

// ---------------------------------------------------------------------------
// LayerNorm (one row per block, 256 threads); writes f32 and/or f16 copy.
// ---------------------------------------------------------------------------
__global__ __launch_bounds__(256) void layernorm_kernel(
    const float* __restrict__ x, const float* __restrict__ g,
    const float* __restrict__ b, float* __restrict__ yf,
    _Float16* __restrict__ yh, int n)
{
  __shared__ float sh1[8], sh2[8];
  long row = blockIdx.x;
  const float* xr = x + row * n;
  float s = 0.f, ss = 0.f;
  for (int i = threadIdx.x; i < n; i += 256) { float v = xr[i]; s += v; ss += v*v; }
  for (int o = 16; o > 0; o >>= 1) { s += __shfl_xor(s, o, 32); ss += __shfl_xor(ss, o, 32); }
  if ((threadIdx.x & 31) == 0) { sh1[threadIdx.x >> 5] = s; sh2[threadIdx.x >> 5] = ss; }
  __syncthreads();
  s = 0.f; ss = 0.f;
  for (int i = 0; i < 8; ++i) { s += sh1[i]; ss += sh2[i]; }
  float mean = s / n;
  float var  = ss / n - mean * mean;
  float inv  = rsqrtf(var + 1e-6f);
  for (int i = threadIdx.x; i < n; i += 256) {
    float v = (xr[i] - mean) * inv * g[i] + b[i];
    if (yf) yf[row * n + i] = v;
    if (yh) yh[row * n + i] = (_Float16)v;
  }
}

// Causal row softmax over scores S[H,L,L] -> probabilities P (f16).
__global__ __launch_bounds__(256) void softmax_causal_kernel(
    const float* __restrict__ S, _Float16* __restrict__ P)
{
  __shared__ float sh[8];
  int i = blockIdx.x, h = blockIdx.y;
  const float* sr = S + ((long)h * Ll + i) * Ll;
  _Float16*    pr = P + ((long)h * Ll + i) * Ll;
  int nvalid = i + 1;
  float mx = -3.0e38f;
  for (int j = threadIdx.x; j < nvalid; j += 256) mx = fmaxf(mx, sr[j]);
  for (int o = 16; o > 0; o >>= 1) mx = fmaxf(mx, __shfl_xor(mx, o, 32));
  if ((threadIdx.x & 31) == 0) sh[threadIdx.x >> 5] = mx;
  __syncthreads();
  mx = sh[0];
  for (int k = 1; k < 8; ++k) mx = fmaxf(mx, sh[k]);
  __syncthreads();
  float sum = 0.f;
  for (int j = threadIdx.x; j < nvalid; j += 256) sum += __expf(sr[j] - mx);
  for (int o = 16; o > 0; o >>= 1) sum += __shfl_xor(sum, o, 32);
  if ((threadIdx.x & 31) == 0) sh[threadIdx.x >> 5] = sum;
  __syncthreads();
  sum = 0.f;
  for (int k = 0; k < 8; ++k) sum += sh[k];
  float inv = 1.0f / sum;
  for (int j = threadIdx.x; j < Ll; j += 256)
    pr[j] = (_Float16)((j < nvalid) ? __expf(sr[j] - mx) * inv : 0.f);
}

// Sparse CDR<-antigen cross attention: one wave per (token, head).
// Lane = neighbor for the score phase, lane = dim-pair for the output phase.
__global__ __launch_bounds__(256) void sparse_cross_attn_kernel(
    const float* __restrict__ Qc, const float* __restrict__ Kc,
    const float* __restrict__ Vc, const int* __restrict__ col_idx,
    float* __restrict__ co_f, _Float16* __restrict__ co_h)
{
  __shared__ float qsh[Dm];
  __shared__ int   nbs[KNBc];
  int l = blockIdx.x;
  for (int i = threadIdx.x; i < Dm; i += 256) qsh[i] = Qc[(long)l * Dm + i];
  if (threadIdx.x < KNBc) nbs[threadIdx.x] = col_idx[l * KNBc + threadIdx.x];
  __syncthreads();
  int h    = threadIdx.x >> 5;     // head (8 waves == 8 heads)
  int lane = threadIdx.x & 31;     // neighbor index (KNB == 32 == wave32)
  int nb   = nbs[lane];
  const float* kr = Kc + (long)nb * Dm + h * DH;
  const float* qh = qsh + h * DH;
  float s = 0.f;
  for (int d = 0; d < DH; ++d) s += qh[d] * kr[d];
  s *= 0.125f;                     // 1/sqrt(64)
  float mx = s;
  for (int o = 16; o > 0; o >>= 1) mx = fmaxf(mx, __shfl_xor(mx, o, 32));
  float e = __expf(s - mx);
  float sum = e;
  for (int o = 16; o > 0; o >>= 1) sum += __shfl_xor(sum, o, 32);
  float p = e / sum;
  // output: each lane owns 2 of the 64 head dims
  float o0 = 0.f, o1 = 0.f;
  for (int k = 0; k < KNBc; ++k) {
    float pk = __shfl(p, k, 32);
    const float* vr = Vc + (long)nbs[k] * Dm + h * DH;
    o0 += pk * vr[lane * 2];
    o1 += pk * vr[lane * 2 + 1];
  }
  long off = (long)l * Dm + h * DH + lane * 2;
  co_f[off]     = o0;  co_f[off + 1] = o1;
  co_h[off]     = (_Float16)o0;
  co_h[off + 1] = (_Float16)o1;
}

// Token embedding + interleaved sin/cos positional encoding.
__global__ __launch_bounds__(256) void embed_pe_kernel(
    const int* __restrict__ tok, const float* __restrict__ emb,
    float* __restrict__ cf, _Float16* __restrict__ ch)
{
  int l = blockIdx.x, p = threadIdx.x;     // p = 0..255 -> dims 2p, 2p+1
  float dv  = __expf((float)(2 * p) * (-9.210340371976184f / (float)Dm));
  float ang = (float)l * dv;
  int t = tok[l];
  float e0 = emb[(long)t * Dm + 2 * p]     + __sinf(ang);
  float e1 = emb[(long)t * Dm + 2 * p + 1] + __cosf(ang);
  long o = (long)l * Dm + 2 * p;
  cf[o] = e0;  cf[o + 1] = e1;
  ch[o] = (_Float16)e0;  ch[o + 1] = (_Float16)e1;
}

// Antigen features [coords(3) | aa_embed(32) | sasa(1)] -> f16 [NAG,36].
__global__ __launch_bounds__(256) void agfeat_kernel(
    const float* __restrict__ coords, const int* __restrict__ types,
    const float* __restrict__ sasa, const float* __restrict__ aa_emb,
    _Float16* __restrict__ x36)
{
  int i = blockIdx.x * 256 + threadIdx.x;
  if (i >= NAGc) return;
  _Float16* o = x36 + (long)i * 36;
  o[0] = (_Float16)coords[i * 3 + 0];
  o[1] = (_Float16)coords[i * 3 + 1];
  o[2] = (_Float16)coords[i * 3 + 2];
  int t = types[i];
  for (int k = 0; k < 32; ++k) o[3 + k] = (_Float16)aa_emb[t * 32 + k];
  o[35] = (_Float16)sasa[i];
}

__global__ __launch_bounds__(256) void gelu_h_kernel(
    const float* __restrict__ x, _Float16* __restrict__ y, long nel)
{
  long i = (long)blockIdx.x * 256 + threadIdx.x;
  if (i < nel) y[i] = (_Float16)gelu_f(x[i]);
}

// Final log_softmax over V=23 columns; one wave per row.
__global__ __launch_bounds__(32) void logsoftmax_kernel(
    const float* __restrict__ lg, float* __restrict__ out)
{
  int i = blockIdx.x, j = threadIdx.x;
  float v = (j < Vv) ? lg[(long)i * Vv + j] : -3.0e38f;
  float mx = v;
  for (int o = 16; o > 0; o >>= 1) mx = fmaxf(mx, __shfl_xor(mx, o, 32));
  float e = (j < Vv) ? __expf(v - mx) : 0.f;
  float sum = e;
  for (int o = 16; o > 0; o >>= 1) sum += __shfl_xor(sum, o, 32);
  if (j < Vv) out[(long)i * Vv + j] = v - mx - __logf(sum);
}

// ---------------------------------------------------------------------------
// Host side
// ---------------------------------------------------------------------------
static void launch_gemm(hipStream_t st, int var,
    const _Float16* A, long lda, long batA,
    const void* B, long ldb, long batB,
    float* Cf, _Float16* Ch, long ldc, long batC,
    const float* bias, const float* resid,
    int M, int N, int K, int batch, float scale)
{
  dim3 g((N + BN - 1) / BN, (M + BM - 1) / BM, batch);
  dim3 blk(256);
  switch (var) {
    case 0: gemm_wmma_kernel<false, false, false><<<g, blk, 0, st>>>(
        A, lda, batA, B, ldb, batB, Cf, Ch, ldc, batC, bias, resid, M, N, K, scale); break;
    case 1: gemm_wmma_kernel<false, false, true ><<<g, blk, 0, st>>>(
        A, lda, batA, B, ldb, batB, Cf, Ch, ldc, batC, bias, resid, M, N, K, scale); break;
    case 2: gemm_wmma_kernel<true,  true,  false><<<g, blk, 0, st>>>(
        A, lda, batA, B, ldb, batB, Cf, Ch, ldc, batC, bias, resid, M, N, K, scale); break;
    case 3: gemm_wmma_kernel<true,  false, false><<<g, blk, 0, st>>>(
        A, lda, batA, B, ldb, batB, Cf, Ch, ldc, batC, bias, resid, M, N, K, scale); break;
  }
}

extern "C" void kernel_launch(void* const* d_in, const int* in_sizes, int n_in,
                              void* d_out, int out_size, void* d_ws, size_t ws_size,
                              hipStream_t stream)
{
  (void)in_sizes; (void)n_in; (void)out_size; (void)ws_size;
  const int*   token_ids   = (const int*)  d_in[0];
  const float* ag_coords   = (const float*)d_in[1];
  const int*   ag_types    = (const int*)  d_in[2];
  const float* ag_sasa     = (const float*)d_in[3];
  const int*   col_idx     = (const int*)  d_in[5];
  const float* tok_embed   = (const float*)d_in[8];
  const float* ag_aa_embed = (const float*)d_in[9];
  const float* enc_W1      = (const float*)d_in[10];
  const float* enc_b1      = (const float*)d_in[11];
  const float* enc_ln_s    = (const float*)d_in[12];
  const float* enc_ln_b    = (const float*)d_in[13];
  const float* enc_W2      = (const float*)d_in[14];
  const float* enc_b2      = (const float*)d_in[15];
  const float* Wq  = (const float*)d_in[16];
  const float* Wk  = (const float*)d_in[17];
  const float* Wv  = (const float*)d_in[18];
  const float* Wo  = (const float*)d_in[19];
  const float* ln1s = (const float*)d_in[20];
  const float* ln1b = (const float*)d_in[21];
  const float* Wqc = (const float*)d_in[22];
  const float* Wkc = (const float*)d_in[23];
  const float* Wvc = (const float*)d_in[24];
  const float* Woc = (const float*)d_in[25];
  const float* ln2s = (const float*)d_in[26];
  const float* ln2b = (const float*)d_in[27];
  const float* fW1 = (const float*)d_in[28];
  const float* fb1 = (const float*)d_in[29];
  const float* fW2 = (const float*)d_in[30];
  const float* fb2 = (const float*)d_in[31];
  const float* ln3s = (const float*)d_in[32];
  const float* ln3b = (const float*)d_in[33];
  const float* out_W = (const float*)d_in[34];
  const float* out_b = (const float*)d_in[35];

  // ---- workspace carve-out ----
  char* w = (char*)d_ws;
  auto take = [&](size_t bytes) {
    void* r = (void*)w;
    w += (bytes + 255) & ~(size_t)255;
    return r;
  };
  float*    cdr_f = (float*)   take((size_t)Ll * Dm * 4);
  _Float16* cdr_h = (_Float16*)take((size_t)Ll * Dm * 2);
  float*    tmp_f = (float*)   take((size_t)Ll * Dm * 4);
  _Float16* q_h   = (_Float16*)take((size_t)Ll * Dm * 2);
  _Float16* k_h   = (_Float16*)take((size_t)Ll * Dm * 2);
  _Float16* v_h   = (_Float16*)take((size_t)Ll * Dm * 2);
  _Float16* o_h   = (_Float16*)take((size_t)Ll * Dm * 2);
  float*    qc_f  = (float*)   take((size_t)Ll * Dm * 4);
  float*    kc_f  = (float*)   take((size_t)NAGc * Dm * 4);
  float*    vc_f  = (float*)   take((size_t)NAGc * Dm * 4);
  float*    co_f  = (float*)   take((size_t)Ll * Dm * 4);
  _Float16* co_h  = (_Float16*)take((size_t)Ll * Dm * 2);
  _Float16* f_h   = (_Float16*)take((size_t)Ll * Ff * 2);
  _Float16* ag_h  = (_Float16*)take((size_t)NAGc * Dm * 2);
  _Float16* x36_h = (_Float16*)take((size_t)NAGc * 36 * 2);
  float*    s_f   = (float*)   take((size_t)Hh * Ll * Ll * 4);
  _Float16* p_h   = (_Float16*)take((size_t)Hh * Ll * Ll * 2);
  float*    logits= (float*)   take((size_t)Ll * Vv * 4);
  // encoder temporaries alias the attention buffers (used strictly before)
  float*    enc_f = s_f;
  _Float16* enc_g = p_h;

  // ---- antigen encoder ----
  agfeat_kernel<<<(NAGc + 255) / 256, 256, 0, stream>>>(
      ag_coords, ag_types, ag_sasa, ag_aa_embed, x36_h);
  launch_gemm(stream, 0, x36_h, 36, 0, enc_W1, Dm, 0,
              enc_f, nullptr, Dm, 0, enc_b1, nullptr, NAGc, Dm, 36, 1, 1.f);
  layernorm_kernel<<<NAGc, 256, 0, stream>>>(enc_f, enc_ln_s, enc_ln_b,
                                             enc_f, nullptr, Dm);
  gelu_h_kernel<<<(unsigned)(((long)NAGc * Dm + 255) / 256), 256, 0, stream>>>(
      enc_f, enc_g, (long)NAGc * Dm);
  launch_gemm(stream, 0, enc_g, Dm, 0, enc_W2, Dm, 0,
              nullptr, ag_h, Dm, 0, enc_b2, nullptr, NAGc, Dm, Dm, 1, 1.f);

  // ---- CDR token embedding + positional encoding ----
  embed_pe_kernel<<<Ll, 256, 0, stream>>>(token_ids, tok_embed, cdr_f, cdr_h);

  // ---- transformer layers ----
  for (int l = 0; l < NLn; ++l) {
    const float* Wq_l  = Wq  + (long)l * Dm * Dm;
    const float* Wk_l  = Wk  + (long)l * Dm * Dm;
    const float* Wv_l  = Wv  + (long)l * Dm * Dm;
    const float* Wo_l  = Wo  + (long)l * Dm * Dm;
    const float* Wqc_l = Wqc + (long)l * Dm * Dm;
    const float* Wkc_l = Wkc + (long)l * Dm * Dm;
    const float* Wvc_l = Wvc + (long)l * Dm * Dm;
    const float* Woc_l = Woc + (long)l * Dm * Dm;
    const float* fW1_l = fW1 + (long)l * Dm * Ff;
    const float* fW2_l = fW2 + (long)l * Ff * Dm;

    // causal self-attention
    launch_gemm(stream, 0, cdr_h, Dm, 0, Wq_l, Dm, 0,
                nullptr, q_h, Dm, 0, nullptr, nullptr, Ll, Dm, Dm, 1, 1.f);
    launch_gemm(stream, 0, cdr_h, Dm, 0, Wk_l, Dm, 0,
                nullptr, k_h, Dm, 0, nullptr, nullptr, Ll, Dm, Dm, 1, 1.f);
    launch_gemm(stream, 0, cdr_h, Dm, 0, Wv_l, Dm, 0,
                nullptr, v_h, Dm, 0, nullptr, nullptr, Ll, Dm, Dm, 1, 1.f);
    // S[h] = Q_h @ K_h^T * 1/sqrt(DH)   (batched over heads, col-sliced)
    launch_gemm(stream, 2, q_h, Dm, DH, k_h, Dm, DH,
                s_f, nullptr, Ll, (long)Ll * Ll, nullptr, nullptr,
                Ll, Ll, DH, Hh, 0.125f);
    softmax_causal_kernel<<<dim3(Ll, Hh), 256, 0, stream>>>(s_f, p_h);
    // O[h] = P_h @ V_h
    launch_gemm(stream, 3, p_h, Ll, (long)Ll * Ll, v_h, Dm, DH,
                nullptr, o_h, Dm, DH, nullptr, nullptr, Ll, DH, Ll, Hh, 1.f);
    launch_gemm(stream, 0, o_h, Dm, 0, Wo_l, Dm, 0,
                tmp_f, nullptr, Dm, 0, nullptr, cdr_f, Ll, Dm, Dm, 1, 1.f);
    layernorm_kernel<<<Ll, 256, 0, stream>>>(tmp_f, ln1s + (long)l * Dm,
                                             ln1b + (long)l * Dm, cdr_f, cdr_h, Dm);

    // sparse cross-attention to antigen
    launch_gemm(stream, 0, cdr_h, Dm, 0, Wqc_l, Dm, 0,
                qc_f, nullptr, Dm, 0, nullptr, nullptr, Ll, Dm, Dm, 1, 1.f);
    launch_gemm(stream, 0, ag_h, Dm, 0, Wkc_l, Dm, 0,
                kc_f, nullptr, Dm, 0, nullptr, nullptr, NAGc, Dm, Dm, 1, 1.f);
    launch_gemm(stream, 0, ag_h, Dm, 0, Wvc_l, Dm, 0,
                vc_f, nullptr, Dm, 0, nullptr, nullptr, NAGc, Dm, Dm, 1, 1.f);
    sparse_cross_attn_kernel<<<Ll, 256, 0, stream>>>(qc_f, kc_f, vc_f,
                                                     col_idx, co_f, co_h);
    launch_gemm(stream, 0, co_h, Dm, 0, Woc_l, Dm, 0,
                tmp_f, nullptr, Dm, 0, nullptr, cdr_f, Ll, Dm, Dm, 1, 1.f);
    layernorm_kernel<<<Ll, 256, 0, stream>>>(tmp_f, ln2s + (long)l * Dm,
                                             ln2b + (long)l * Dm, cdr_f, cdr_h, Dm);

    // FFN
    launch_gemm(stream, 1, cdr_h, Dm, 0, fW1_l, Ff, 0,
                nullptr, f_h, Ff, 0, fb1 + (long)l * Ff, nullptr,
                Ll, Ff, Dm, 1, 1.f);
    launch_gemm(stream, 0, f_h, Ff, 0, fW2_l, Dm, 0,
                tmp_f, nullptr, Dm, 0, fb2 + (long)l * Dm, cdr_f,
                Ll, Dm, Ff, 1, 1.f);
    layernorm_kernel<<<Ll, 256, 0, stream>>>(tmp_f, ln3s + (long)l * Dm,
                                             ln3b + (long)l * Dm, cdr_f, cdr_h, Dm);
  }

  // ---- output head ----
  launch_gemm(stream, 0, cdr_h, Dm, 0, out_W, Vv, 0,
              logits, nullptr, Vv, 0, out_b, nullptr, Ll, Vv, Dm, 1, 1.f);
  logsoftmax_kernel<<<Ll, 32, 0, stream>>>(logits, (float*)d_out);
}
